// DeepseekV4MLAAttention_58566174048947
// MI455X (gfx1250) — matrix operand order repaired and checked by reading, not compiled, not run on hardware
//
#include <hip/hip_runtime.h>

typedef __attribute__((ext_vector_type(16))) __bf16 v16bf;
typedef __attribute__((ext_vector_type(8)))  __bf16 v8bf;
typedef __attribute__((ext_vector_type(4)))  __bf16 v4bf;
typedef __attribute__((ext_vector_type(8)))  float  v8f;

#define NTOK  512
#define NH    64
#define DK    576
#define DV    512
#define TOPK  512
#define KT    64                        // key tile (4x4 = 16 score tiles)
#define NTILE (TOPK / KT)               // 8
#define NTHR  1024                      // 32 waves
#define CHUNKS (KT * (DK / 4) / NTHR)   // 9 8-byte chunks per thread per tile
#define SCALEF 0.04166666666666667f     // 576^-0.5 = 1/24

// ---- padded LDS pitches (elements) to break bank conflicts ----
#define QP 584   // Q row pitch (bf16): 1168B = 292 words; 292%64=36, gcd 4
#define KP 584   // K key-major row pitch (bf16)
#define VP 68    // V d-major pitch (bf16): 136B; 4-row store stride 136w%64=8
#define SP 65    // S row pitch (f32): gcd(65,64)=1 -> conflict-free
#define PP 72    // P row pitch (bf16): 144B, 16B aligned

// ---- LDS layout (dynamic shared) ----
#define OFF_Q    0                      // 64*584*2  = 74752
#define OFF_K    74752                  // 64*584*2  = 74752
#define OFF_V    149504                 // 512*68*2  = 69632
#define OFF_S    219136                 // 64*65*4   = 16640
#define OFF_P    235776                 // 64*72*2   = 9216
#define OFF_M    244992
#define OFF_L    245248
#define OFF_A    245504
#define OFF_KADD 245760                 // 64*4
#define SMEM_BYTES 246016

__global__ void cvt_kv_bf16(const float* __restrict__ kv, __bf16* __restrict__ dst, int n) {
  int i = (blockIdx.x * blockDim.x + threadIdx.x) * 4;
  int stride = gridDim.x * blockDim.x * 4;
  for (; i < n; i += stride) {
    float4 v = *(const float4*)(kv + i);
    dst[i + 0] = (__bf16)v.x;
    dst[i + 1] = (__bf16)v.y;
    dst[i + 2] = (__bf16)v.z;
    dst[i + 3] = (__bf16)v.w;
  }
}

__launch_bounds__(NTHR, 1)
__global__ void mla_sparse_attn(const float* __restrict__ q,
                                const __bf16* __restrict__ kvbf,
                                const int* __restrict__ topk,
                                const float* __restrict__ sink,
                                float* __restrict__ out) {
  extern __shared__ char smem[];
  __bf16* Qs   = (__bf16*)(smem + OFF_Q);
  __bf16* Kkm  = (__bf16*)(smem + OFF_K);
  __bf16* Vdm  = (__bf16*)(smem + OFF_V);
  float*  Sbuf = (float*) (smem + OFF_S);
  __bf16* Pbuf = (__bf16*)(smem + OFF_P);
  float*  mrun = (float*) (smem + OFF_M);
  float*  lrun = (float*) (smem + OFF_L);
  float*  alph = (float*) (smem + OFF_A);
  float*  kadd = (float*) (smem + OFF_KADD);

  const int tid  = threadIdx.x;
  const int wv   = tid >> 5;           // wave id 0..31
  const int lane = tid & 31;
  const int hf   = lane >> 4;          // lane half (0/1) per WMMA layouts
  const int lm   = lane & 15;
  const int t    = blockIdx.x;

  // score phase (waves 0..15): h-tile = (wv&3), key n-tile = (wv>>2)
  // out phase (all 32 waves): 16 heads x 64 output cols per wave
  const int hBase = (wv & 3) * 16;
  const int nBase = ((wv >> 2) & 3) * 16;
  const int dBase = (wv >> 2) * 64;    // 4 d-tiles of 16

  // ---- stage q[t] as bf16 (padded pitch) ----
  const float* qrow = q + (size_t)t * NH * DK;
  for (int i = tid; i < NH * DK; i += NTHR)
    Qs[(i / DK) * QP + (i % DK)] = (__bf16)qrow[i];

  // ---- init online softmax state: fold sink into (m=sink, l=1) ----
  if (tid < NH) {
    mrun[tid] = sink[tid];
    lrun[tid] = 1.0f;
  }

  v8f acc[4];
  for (int dt = 0; dt < 4; ++dt)
    for (int e = 0; e < 8; ++e) acc[dt][e] = 0.0f;

  const int* tkrow = topk + (size_t)t * TOPK;

  // ---- register double-buffer for the gather: CHUNKS x 8B per thread ----
  uint2 rbuf[CHUNKS];
#pragma unroll
  for (int j = 0; j < CHUNKS; ++j) {
    int c = tid + j * NTHR;
    int key = c / (DK / 4);
    int off4 = (c % (DK / 4)) * 4;
    int idx = tkrow[key];
    int row = (idx < 0) ? 0 : idx;
    rbuf[j] = *(const uint2*)(kvbf + (size_t)row * DK + off4);
  }

  for (int tile = 0; tile < NTILE; ++tile) {
    __syncthreads();   // previous iteration's LDS consumers done

    // per-key valid flags
    if (tid < KT)
      kadd[tid] = (tkrow[tile * KT + tid] < 0) ? -__builtin_inff() : 0.0f;

    // ---- commit prefetched tile to LDS: Kkm (key-major) + Vdm (d-major) ----
#pragma unroll
    for (int j = 0; j < CHUNKS; ++j) {
      int c = tid + j * NTHR;
      int key = c / (DK / 4);
      int off4 = (c % (DK / 4)) * 4;
      *(uint2*)(Kkm + key * KP + off4) = rbuf[j];
      if (off4 < DV) {
        const __bf16* s4 = (const __bf16*)&rbuf[j];
        Vdm[(off4 + 0) * VP + key] = s4[0];
        Vdm[(off4 + 1) * VP + key] = s4[1];
        Vdm[(off4 + 2) * VP + key] = s4[2];
        Vdm[(off4 + 3) * VP + key] = s4[3];
      }
    }

    // ---- issue next tile's global gather; completes under the compute ----
    if (tile + 1 < NTILE) {
#pragma unroll
      for (int j = 0; j < CHUNKS; ++j) {
        int c = tid + j * NTHR;
        int key = c / (DK / 4);
        int off4 = (c % (DK / 4)) * 4;
        int idx = tkrow[(tile + 1) * KT + key];
        int row = (idx < 0) ? 0 : idx;
        rbuf[j] = *(const uint2*)(kvbf + (size_t)row * DK + off4);
      }
    }
    __syncthreads();

    // ---- scores: waves 0..15, one 16x16 tile each (4 h-tiles x 4 n-tiles) ----
    if (wv < 16) {
      v8f c;
      for (int e = 0; e < 8; ++e) c[e] = 0.0f;
      const __bf16* qb = Qs  + (hBase + lm) * QP;
      const __bf16* kb = Kkm + (nBase + lm) * KP;
      for (int d0 = 0; d0 < DK; d0 += 32) {
        union { v16bf v; v8bf h[2]; } ua, ub;
        // A (16x32 bf16): lane=row, elems = {K[b..b+7], K[b+16..b+23]}, b=hf*8
        ua.h[0] = *(const v8bf*)(qb + d0 + hf * 8);
        ua.h[1] = *(const v8bf*)(qb + d0 + hf * 8 + 16);
        // B (32x16 bf16): lane=col, elems = {K[b..b+7], K[b+16..b+23]}, b=hf*8
        ub.h[0] = *(const v8bf*)(kb + d0 + hf * 8);
        ub.h[1] = *(const v8bf*)(kb + d0 + hf * 8 + 16);
        c = __builtin_amdgcn_wmma_f32_16x16x32_bf16(false, ua.v, false, ub.v,
                                                    (short)0, c, false, false);
      }
      // C layout: VGPR r, lane -> (row = r + 8*hf, col = lm)
      for (int r = 0; r < 8; ++r)
        Sbuf[(hBase + r + 8 * hf) * SP + nBase + lm] = c[r];
    }
    __syncthreads();

    // ---- online softmax, one thread per head (two conflict-free LDS passes) ----
    if (tid < NH) {
      float m0 = mrun[tid];
      float tmax = -__builtin_inff();
      for (int k = 0; k < KT; ++k)
        tmax = fmaxf(tmax, Sbuf[tid * SP + k] * SCALEF + kadd[k]);
      float mn = fmaxf(m0, tmax);
      float al = __expf(m0 - mn);
      float sum = 0.0f;
      for (int k = 0; k < KT; ++k) {
        float p = __expf(Sbuf[tid * SP + k] * SCALEF + kadd[k] - mn);
        sum += p;
        Pbuf[tid * PP + k] = (__bf16)p;
      }
      lrun[tid] = lrun[tid] * al + sum;
      mrun[tid] = mn;
      alph[tid] = al;
    }
    __syncthreads();

    // ---- rescale accumulators and accumulate P x V (K=64, kk-outer) ----
    {
      float al8[8];
      for (int e = 0; e < 8; ++e) al8[e] = alph[hBase + e + 8 * hf];
      for (int dt = 0; dt < 4; ++dt)
        for (int e = 0; e < 8; ++e) acc[dt][e] *= al8[e];

      const __bf16* pb = Pbuf + (hBase + lm) * PP;
#pragma unroll
      for (int kk = 0; kk < 2; ++kk) {
        union { v16bf v; v8bf h[2]; } ua;
        ua.h[0] = *(const v8bf*)(pb + kk * 32 + hf * 8);
        ua.h[1] = *(const v8bf*)(pb + kk * 32 + hf * 8 + 16);
        for (int dt = 0; dt < 4; ++dt) {
          int n0 = dBase + dt * 16;
          union { v16bf v; v4bf qv[4]; } ub;
          const __bf16* vb = Vdm + (n0 + lm) * VP + kk * 32 + hf * 8;
          ub.qv[0] = *(const v4bf*)(vb + 0);
          ub.qv[1] = *(const v4bf*)(vb + 4);
          ub.qv[2] = *(const v4bf*)(vb + 16);
          ub.qv[3] = *(const v4bf*)(vb + 20);
          acc[dt] = __builtin_amdgcn_wmma_f32_16x16x32_bf16(false, ua.v, false, ub.v,
                                                            (short)0, acc[dt], false, false);
        }
      }
    }
  }

  __syncthreads();

  // ---- normalize and store: out[t][h][d] f32 ----
  {
    float inv8[8];
    for (int e = 0; e < 8; ++e) inv8[e] = 1.0f / lrun[hBase + e + 8 * hf];
    float* ob = out + (size_t)t * NH * DV;
    for (int dt = 0; dt < 4; ++dt) {
      int dcol = dBase + dt * 16 + lm;
      for (int e = 0; e < 8; ++e) {
        int h = hBase + e + 8 * hf;
        ob[h * DV + dcol] = acc[dt][e] * inv8[e];
      }
    }
  }
}

extern "C" void kernel_launch(void* const* d_in, const int* in_sizes, int n_in,
                              void* d_out, int out_size, void* d_ws, size_t ws_size,
                              hipStream_t stream) {
  const float* q     = (const float*)d_in[0];
  const float* kv    = (const float*)d_in[1];
  const int*   topk  = (const int*)d_in[2];
  const float* sink  = (const float*)d_in[3];
  float*       out   = (float*)d_out;
  __bf16*      kvbf  = (__bf16*)d_ws;   // 32768*576*2 = 37.75 MB scratch

  int nkv = in_sizes[1];                // 32768*576, multiple of 4
  int blocks = (nkv / 4 + 255) / 256;
  cvt_kv_bf16<<<blocks, 256, 0, stream>>>(kv, kvbf, nkv);

  (void)hipFuncSetAttribute((const void*)mla_sparse_attn,
                            hipFuncAttributeMaxDynamicSharedMemorySize, SMEM_BYTES);
  mla_sparse_attn<<<NTOK, NTHR, SMEM_BYTES, stream>>>(q, kvbf, topk, sink, out);
}
